// TransformerLayer_45603962749382
// MI455X (gfx1250) — matrix operand order repaired and checked by reading, not compiled, not run on hardware
//
#include <hip/hip_runtime.h>
#include <hip/hip_bf16.h>
#include <math.h>

#define DEV_INLINE __device__ __forceinline__

constexpr int Bc   = 4;
constexpr int Nn   = 2048;
constexpr int Dd   = 256;
constexpr int Hh   = 4;
constexpr int HDc  = 64;
constexpr int Rows = Bc * Nn;   // 8192
constexpr int BHc  = Bc * Hh;   // 16

typedef __attribute__((ext_vector_type(16))) __bf16 v16bf;
typedef __attribute__((ext_vector_type(8)))  float  v8f;

union FragAB { unsigned u[8]; v16bf v; };
union FragC  { float   f[8]; v8f   v; };

DEV_INLINE unsigned short f2bf_bits(float x) {
  unsigned u = __float_as_uint(x);
  u += 0x7fffu + ((u >> 16) & 1u);           // round-to-nearest-even
  return (unsigned short)(u >> 16);
}
DEV_INLINE __bf16 f2bf(float x) {
  unsigned short h = f2bf_bits(x);
  __bf16 r;
  __builtin_memcpy(&r, &h, 2);
  return r;
}

DEV_INLINE v8f wmma_bf16(v16bf a, v16bf b, v8f c) {
  // D = A(16x32 bf16) * B(32x16 bf16) + C(16x16 f32)
  return __builtin_amdgcn_wmma_f32_16x16x32_bf16(false, a, false, b, (short)0, c,
                                                 false, false);
}

// ---- CDNA5 async global->LDS staging (guarded; falls back to uint4 copies) ----
#if defined(__has_builtin)
#  if __has_builtin(__builtin_amdgcn_global_load_async_to_lds_b128)
#    define USE_ASYNC_LDS 1
#  endif
#endif

#ifdef USE_ASYNC_LDS
typedef int async_v4i __attribute__((vector_size(4 * sizeof(int))));
DEV_INLINE void async_copy16(const void* g, void* l) {
  __builtin_amdgcn_global_load_async_to_lds_b128(
      (__attribute__((address_space(1))) async_v4i*)g,
      (__attribute__((address_space(3))) async_v4i*)l, 0, 0);
}
DEV_INLINE void async_wait0() {
#  if __has_builtin(__builtin_amdgcn_s_wait_asynccnt)
  __builtin_amdgcn_s_wait_asynccnt(0);
#  else
  asm volatile("s_wait_asynccnt 0x0" ::: "memory");
#  endif
}
#endif

// ---------------------------------------------------------------------------
// Generic GEMM:  C[M,N] = A[M,K](bf16) * W[N,K](bf16)^T + bias[N] (+ resid[M,N])
// block = 128 threads (4 waves), tile 128(M) x 64(N), k-step 32.
// Each wave computes 32x64: 2 A-frags x 4 B-frags = 8 WMMA per k-step,
// with each B fragment reused for both A fragments.
// ---------------------------------------------------------------------------
__global__ __launch_bounds__(128)
void gemm_bf16_kernel(const __bf16* __restrict__ A,
                      const __bf16* __restrict__ W,
                      const float*  __restrict__ bias,
                      const float*  __restrict__ resid,
                      float*        __restrict__ C,
                      int M, int K, int N)
{
  __shared__ __align__(16) __bf16 sA[128 * 32];   // 8 KB
  __shared__ __align__(16) __bf16 sB[64 * 32];    // 4 KB

  const int tid  = threadIdx.x;
  const int wv   = tid >> 5;
  const int lane = tid & 31;
  const int hl   = lane >> 4;
  const int lm   = lane & 15;
  const int m0   = blockIdx.x * 128;
  const int n0   = blockIdx.y * 64;

  // hoisted staging addresses: thread owns (row = tid/4 + j*32, chunk = tid%4)
  const int arow = tid >> 2;
  const int ach  = tid & 3;
  const __bf16* pa = A + (size_t)(m0 + arow) * K + ach * 8;
  const __bf16* pb = W + (size_t)(n0 + arow) * K + ach * 8;
  const size_t rstep = (size_t)32 * K;            // 32 rows in elements

  FragC acc[2][4];
#pragma unroll
  for (int mh = 0; mh < 2; ++mh)
#pragma unroll
    for (int t = 0; t < 4; ++t)
#pragma unroll
      for (int r = 0; r < 8; ++r) acc[mh][t].f[r] = 0.f;

  for (int k0 = 0; k0 < K; k0 += 32) {
#ifdef USE_ASYNC_LDS
#pragma unroll
    for (int j = 0; j < 4; ++j)
      async_copy16(pa + j * rstep, &sA[(arow + j * 32) * 32 + ach * 8]);
#pragma unroll
    for (int j = 0; j < 2; ++j)
      async_copy16(pb + j * rstep, &sB[(arow + j * 32) * 32 + ach * 8]);
    async_wait0();
#else
#pragma unroll
    for (int j = 0; j < 4; ++j)
      *reinterpret_cast<uint4*>(&sA[(arow + j * 32) * 32 + ach * 8]) =
          *reinterpret_cast<const uint4*>(pa + j * rstep);
#pragma unroll
    for (int j = 0; j < 2; ++j)
      *reinterpret_cast<uint4*>(&sB[(arow + j * 32) * 32 + ach * 8]) =
          *reinterpret_cast<const uint4*>(pb + j * rstep);
    __builtin_prefetch(pa + 32, 0, 3);
    __builtin_prefetch(pb + 32, 0, 3);
#endif
    pa += 32;
    pb += 32;
    __syncthreads();

    // A fragments: lane -> rows M = wv*32 + mh*16 + lm; pair p covers
    // K = (p>=4 ? 16 : 0) + 8*half + 2*(p&3)
    FragAB fa[2];
#pragma unroll
    for (int mh = 0; mh < 2; ++mh)
#pragma unroll
      for (int p = 0; p < 8; ++p) {
        const int koff = ((p >> 2) << 4) + (hl << 3) + ((p & 3) << 1);
        fa[mh].u[p] = *reinterpret_cast<const unsigned*>(
            &sA[(wv * 32 + mh * 16 + lm) * 32 + koff]);
      }
#pragma unroll
    for (int t = 0; t < 4; ++t) {
      // B fragment: lane -> col N = t*16 + lm; pair p covers K = (p>=4 ? 16 : 0) + 8*half + 2*(p&3)  (reused twice)
      FragAB fb;
#pragma unroll
      for (int p = 0; p < 8; ++p)
        fb.u[p] = *reinterpret_cast<const unsigned*>(
            &sB[(t * 16 + lm) * 32 + ((p >> 2) << 4) + (hl << 3) + ((p & 3) << 1)]);
      acc[0][t].v = wmma_bf16(fa[0].v, fb.v, acc[0][t].v);
      acc[1][t].v = wmma_bf16(fa[1].v, fb.v, acc[1][t].v);
    }
    __syncthreads();
  }

  float biasn[4];
#pragma unroll
  for (int t = 0; t < 4; ++t) biasn[t] = bias[n0 + t * 16 + lm];

#pragma unroll
  for (int mh = 0; mh < 2; ++mh)
#pragma unroll
    for (int t = 0; t < 4; ++t)
#pragma unroll
      for (int r = 0; r < 8; ++r) {
        const int m = m0 + wv * 32 + mh * 16 + hl * 8 + r;
        const int n = n0 + t * 16 + lm;
        float v = acc[mh][t].f[r] + biasn[t];
        if (resid) v += resid[(size_t)m * N + n];
        C[(size_t)m * N + n] = v;
      }
}

// ---------------------------------------------------------------------------
// Flash attention: softmax over keys.  Q,K,V : [BH][N][64] bf16 (scales pre-folded).
// 1 wave per 16 query rows. ctx out: [B][N][256] bf16, channel = h*64 + f.
// ---------------------------------------------------------------------------
__global__ __launch_bounds__(32)
void attn_kernel(const __bf16* __restrict__ Q,
                 const __bf16* __restrict__ Km,
                 const __bf16* __restrict__ V,
                 __bf16*       __restrict__ ctx)
{
  __shared__ __align__(16) __bf16 sP[16 * 32];   // P tile, C-layout -> A-layout transpose
  __shared__ __align__(16) __bf16 sV[32 * 64];   // staged V chunk

  const int lane = threadIdx.x;
  const int hl   = lane >> 4;
  const int lm   = lane & 15;
  const int qt   = blockIdx.x;   // query tile (N/16)
  const int bh   = blockIdx.y;   // batch*head

  const __bf16* qbase = Q + ((size_t)bh * Nn + qt * 16) * HDc;

  // preload Q fragments (2 k-steps over HD=64)
  FragAB fq[2];
#pragma unroll
  for (int ks = 0; ks < 2; ++ks)
#pragma unroll
    for (int p = 0; p < 8; ++p) {
      const int f = ks * 32 + ((p >> 2) << 4) + (hl << 3) + ((p & 3) << 1);
      fq[ks].u[p] = *reinterpret_cast<const unsigned*>(qbase + (size_t)lm * HDc + f);
    }

  FragC o[4];
#pragma unroll
  for (int t = 0; t < 4; ++t)
#pragma unroll
    for (int r = 0; r < 8; ++r) o[t].f[r] = 0.f;

  float mrow[8], lrow[8];
#pragma unroll
  for (int r = 0; r < 8; ++r) { mrow[r] = -3.0e38f; lrow[r] = 0.f; }

  for (int c0 = 0; c0 < Nn; c0 += 32) {
    // ---- S = Q * K^T for two 16-key tiles (global reads only) ----
    FragC s[2];
#pragma unroll
    for (int t2 = 0; t2 < 2; ++t2) {
#pragma unroll
      for (int r = 0; r < 8; ++r) s[t2].f[r] = 0.f;
      const __bf16* kb = Km + ((size_t)bh * Nn + c0 + t2 * 16 + lm) * HDc;
#pragma unroll
      for (int ks = 0; ks < 2; ++ks) {
        FragAB fk;
#pragma unroll
        for (int p = 0; p < 8; ++p)
          fk.u[p] = *reinterpret_cast<const unsigned*>(kb + ks * 32 + ((p >> 2) << 4) + (hl << 3) + ((p & 3) << 1));
        s[t2].v = wmma_bf16(fq[ks].v, fk.v, s[t2].v);
      }
    }

    // ---- online softmax (rows m = 8*hl + r, uniform across the 16 N-lanes) ----
    float rmax[8];
#pragma unroll
    for (int r = 0; r < 8; ++r) rmax[r] = fmaxf(s[0].f[r], s[1].f[r]);
#pragma unroll
    for (int d = 1; d < 16; d <<= 1)
#pragma unroll
      for (int r = 0; r < 8; ++r) rmax[r] = fmaxf(rmax[r], __shfl_xor(rmax[r], d, 32));

    float sc[8], rsum[8];
#pragma unroll
    for (int r = 0; r < 8; ++r) {
      const float nm = fmaxf(mrow[r], rmax[r]);
      sc[r]   = __expf(mrow[r] - nm);
      mrow[r] = nm;
      rsum[r] = 0.f;
    }

    __syncthreads();   // previous iteration's sP/sV reads complete
#pragma unroll
    for (int t2 = 0; t2 < 2; ++t2)
#pragma unroll
      for (int r = 0; r < 8; ++r) {
        const float e = __expf(s[t2].f[r] - mrow[r]);
        rsum[r] += e;
        sP[(hl * 8 + r) * 32 + t2 * 16 + lm] = f2bf(e);
      }
#pragma unroll
    for (int d = 1; d < 16; d <<= 1)
#pragma unroll
      for (int r = 0; r < 8; ++r) rsum[r] += __shfl_xor(rsum[r], d, 32);
#pragma unroll
    for (int r = 0; r < 8; ++r) lrow[r] = lrow[r] * sc[r] + rsum[r];
#pragma unroll
    for (int t = 0; t < 4; ++t)
#pragma unroll
      for (int r = 0; r < 8; ++r) o[t].f[r] *= sc[r];

    // ---- stage V chunk: 32 keys x 64 feats (4KB) ----
    const __bf16* vsrc = V + ((size_t)bh * Nn + c0) * HDc;
#ifdef USE_ASYNC_LDS
#pragma unroll
    for (int j = 0; j < 8; ++j) {
      const int i = lane + j * 32;
      async_copy16(vsrc + i * 8, &sV[i * 8]);
    }
    async_wait0();
#else
    for (int i = lane; i < 256; i += 32)
      *reinterpret_cast<uint4*>(&sV[i * 8]) =
          *reinterpret_cast<const uint4*>(vsrc + i * 8);
#endif
    __syncthreads();

    // ---- O += P * V ----
    FragAB fp;
#pragma unroll
    for (int p = 0; p < 8; ++p) {
      const int koff = ((p >> 2) << 4) + (hl << 3) + ((p & 3) << 1);
      fp.u[p] = *reinterpret_cast<const unsigned*>(&sP[lm * 32 + koff]);
    }
#pragma unroll
    for (int t = 0; t < 4; ++t) {
      FragAB fv;
#pragma unroll
      for (int p = 0; p < 8; ++p) {
        const int key = ((p >> 2) << 4) + (hl << 3) + ((p & 3) << 1);
        const unsigned lo = *reinterpret_cast<const unsigned short*>(&sV[key * 64 + t * 16 + lm]);
        const unsigned hi = *reinterpret_cast<const unsigned short*>(&sV[(key + 1) * 64 + t * 16 + lm]);
        fv.u[p] = lo | (hi << 16);
      }
      o[t].v = wmma_bf16(fp.v, fv.v, o[t].v);
    }
  }

  // epilogue: normalize and scatter to [b][n][h*64+f]
  const int b = bh >> 2, h = bh & 3;
#pragma unroll
  for (int t = 0; t < 4; ++t)
#pragma unroll
    for (int r = 0; r < 8; ++r) {
      const int m = hl * 8 + r;
      const float val = o[t].f[r] / lrow[r];
      ctx[((size_t)b * Nn + qt * 16 + m) * Dd + h * HDc + t * 16 + lm] = f2bf(val);
    }
}

// ---------------------------------------------------------------------------
// Elementwise kernels
// ---------------------------------------------------------------------------
__global__ void f2bf_kernel(const float* __restrict__ in, __bf16* __restrict__ out, int n) {
  const int i = (blockIdx.x * 256 + threadIdx.x) * 4;
  if (i + 3 < n) {
    const float4 v = *reinterpret_cast<const float4*>(in + i);
    out[i]     = f2bf(v.x);
    out[i + 1] = f2bf(v.y);
    out[i + 2] = f2bf(v.z);
    out[i + 3] = f2bf(v.w);
  } else {
    for (int j = i; j < n; ++j) out[j] = f2bf(in[j]);
  }
}

// qkv f32 [8192][768] (channel = h*192 + d*3 + {q,k,v}); enc f32 [2][B][1][N][64]
// -> q,k,v bf16 [BH][2048][64]; RoPE on q,k; qscale folded into q.
__global__ __launch_bounds__(256)
void rope_pack_kernel(const float* __restrict__ qkv,
                      const float* __restrict__ enc,
                      __bf16* __restrict__ qo, __bf16* __restrict__ ko,
                      __bf16* __restrict__ vo, float qscale)
{
  const int rowN = blockIdx.x;          // b*2048 + n
  const int t    = threadIdx.x;         // h*64 + d
  const int h = t >> 6, d = t & 63;
  const int b = rowN >> 11, n = rowN & 2047;

  const float* base = qkv + (size_t)rowN * 768 + h * 192;
  const float q = base[d * 3 + 0];
  const float k = base[d * 3 + 1];
  const float v = base[d * 3 + 2];
  const int dp = (d & 1) ? (d - 1) : (d + 1);
  const float qp = base[dp * 3 + 0];
  const float kp = base[dp * 3 + 1];
  const float rq = (d & 1) ? qp : -qp;    // rotate_half
  const float rk = (d & 1) ? kp : -kp;

  const size_t eoff = ((size_t)b * Nn + n) * HDc + d;
  const float cs = enc[eoff];
  const float sn = enc[(size_t)Bc * Nn * HDc + eoff];

  const size_t oidx = ((size_t)(b * Hh + h) * Nn + n) * HDc + d;
  qo[oidx] = f2bf((q * cs + rq * sn) * qscale);
  ko[oidx] = f2bf(k * cs + rk * sn);
  vo[oidx] = f2bf(v);
}

// f32 [8192][256] (channel = h*64+d) -> bf16 [BH][2048][64], scaled
__global__ __launch_bounds__(256)
void pack_scale_kernel(const float* __restrict__ in, __bf16* __restrict__ out, float scale)
{
  const int rowN = blockIdx.x;
  const int t    = threadIdx.x;
  const int h = t >> 6, d = t & 63;
  const int b = rowN >> 11, n = rowN & 2047;
  out[((size_t)(b * Hh + h) * Nn + n) * HDc + d] =
      f2bf(in[(size_t)rowN * Dd + h * HDc + d] * scale);
}

// concat([x (f32 [rows][256]), m (f32 [rows][256])]) -> bf16 [rows][512]
__global__ void concat_kernel(const float* __restrict__ x, const float* __restrict__ m,
                              __bf16* __restrict__ out, int rows)
{
  const int i = blockIdx.x * 256 + threadIdx.x;
  if (i >= rows * 512) return;
  const int r = i >> 9, c = i & 511;
  const float v = (c < 256) ? x[(size_t)r * 256 + c] : m[(size_t)r * 256 + (c - 256)];
  out[i] = f2bf(v);
}

// LayerNorm(512) + exact GELU, f32 in -> bf16 out. One row per block.
__global__ __launch_bounds__(256)
void ln_gelu_kernel(const float* __restrict__ X, const float* __restrict__ g,
                    const float* __restrict__ be, __bf16* __restrict__ Y)
{
  __shared__ float red[256];
  const int row = blockIdx.x;
  const int t   = threadIdx.x;
  const float* x = X + (size_t)row * 512;
  const float a = x[t], b = x[t + 256];

  red[t] = a + b;
  __syncthreads();
  for (int s = 128; s > 0; s >>= 1) { if (t < s) red[t] += red[t + s]; __syncthreads(); }
  const float mean = red[0] * (1.f / 512.f);
  __syncthreads();

  const float da = a - mean, db = b - mean;
  red[t] = da * da + db * db;
  __syncthreads();
  for (int s = 128; s > 0; s >>= 1) { if (t < s) red[t] += red[t + s]; __syncthreads(); }
  const float rstd = rsqrtf(red[0] * (1.f / 512.f) + 1e-5f);

  float y0 = da * rstd * g[t] + be[t];
  float y1 = db * rstd * g[t + 256] + be[t + 256];
  y0 = 0.5f * y0 * (1.f + erff(y0 * 0.70710678118654752f));
  y1 = 0.5f * y1 * (1.f + erff(y1 * 0.70710678118654752f));
  __bf16* y = Y + (size_t)row * 512;
  y[t]       = f2bf(y0);
  y[t + 256] = f2bf(y1);
}

// ---------------------------------------------------------------------------
// Host orchestration
// ---------------------------------------------------------------------------
static inline void conv_f2bf(hipStream_t s, const float* src, __bf16* dst, int n) {
  f2bf_kernel<<<dim3((n + 1023) / 1024), dim3(256), 0, s>>>(src, dst, n);
}
static inline void gemm(hipStream_t s, const __bf16* A, const __bf16* W, const float* bias,
                        const float* resid, float* C, int M, int K, int N) {
  gemm_bf16_kernel<<<dim3(M / 128, N / 64), dim3(128), 0, s>>>(A, W, bias, resid, C, M, K, N);
}

extern "C" void kernel_launch(void* const* d_in, const int* in_sizes, int n_in,
                              void* d_out, int out_size, void* d_ws, size_t ws_size,
                              hipStream_t stream)
{
  (void)in_sizes; (void)n_in; (void)out_size; (void)ws_size;

  const float* desc0  = (const float*)d_in[0];
  const float* desc1  = (const float*)d_in[1];
  const float* enc0   = (const float*)d_in[2];
  const float* enc1   = (const float*)d_in[3];
  const float* s_Wqkv = (const float*)d_in[4];
  const float* s_bqkv = (const float*)d_in[5];
  const float* s_Wout = (const float*)d_in[6];
  const float* s_bout = (const float*)d_in[7];
  const float* s_W1   = (const float*)d_in[8];
  const float* s_b1   = (const float*)d_in[9];
  const float* s_g    = (const float*)d_in[10];
  const float* s_be   = (const float*)d_in[11];
  const float* s_W2   = (const float*)d_in[12];
  const float* s_b2   = (const float*)d_in[13];
  const float* c_Wqk  = (const float*)d_in[14];
  const float* c_bqk  = (const float*)d_in[15];
  const float* c_Wv   = (const float*)d_in[16];
  const float* c_bv   = (const float*)d_in[17];
  const float* c_Wo   = (const float*)d_in[18];
  const float* c_bo   = (const float*)d_in[19];
  const float* c_W1   = (const float*)d_in[20];
  const float* c_b1   = (const float*)d_in[21];
  const float* c_g    = (const float*)d_in[22];
  const float* c_be   = (const float*)d_in[23];
  const float* c_W2   = (const float*)d_in[24];
  const float* c_b2   = (const float*)d_in[25];

  char* ws = (char*)d_ws;
  size_t off = 0;
  auto alloc = [&](size_t bytes) -> char* {
    char* p = ws + off;
    off += (bytes + 255) & ~(size_t)255;
    return p;
  };

  __bf16* w_qkv = (__bf16*)alloc((size_t)768 * 256 * 2);
  __bf16* w_out = (__bf16*)alloc((size_t)256 * 256 * 2);
  __bf16* w_w1  = (__bf16*)alloc((size_t)512 * 512 * 2);
  __bf16* w_w2  = (__bf16*)alloc((size_t)256 * 512 * 2);
  __bf16* w_cqk = (__bf16*)alloc((size_t)256 * 256 * 2);
  __bf16* w_cv  = (__bf16*)alloc((size_t)256 * 256 * 2);
  __bf16* w_co  = (__bf16*)alloc((size_t)256 * 256 * 2);
  __bf16* w_cw1 = (__bf16*)alloc((size_t)512 * 512 * 2);
  __bf16* w_cw2 = (__bf16*)alloc((size_t)256 * 512 * 2);

  __bf16* aA  = (__bf16*)alloc((size_t)Rows * 512 * 2);     // generic bf16 A operand
  float*  g0  = (float*)alloc((size_t)Rows * 768 * 4);      // generic f32 GEMM out
  __bf16* qb  = (__bf16*)alloc((size_t)BHc * Nn * HDc * 2);
  __bf16* kb  = (__bf16*)alloc((size_t)BHc * Nn * HDc * 2);
  __bf16* vb  = (__bf16*)alloc((size_t)BHc * Nn * HDc * 2);
  __bf16* vb2 = (__bf16*)alloc((size_t)BHc * Nn * HDc * 2);
  float*  msg = (float*)alloc((size_t)Rows * Dd * 4);
  float*  d0s = (float*)alloc((size_t)Rows * Dd * 4);
  float*  d1s = (float*)alloc((size_t)Rows * Dd * 4);
  __bf16* db0 = (__bf16*)alloc((size_t)Rows * Dd * 2);
  __bf16* db1 = (__bf16*)alloc((size_t)Rows * Dd * 2);

  // ---- convert weights to bf16 ----
  conv_f2bf(stream, s_Wqkv, w_qkv, 768 * 256);
  conv_f2bf(stream, s_Wout, w_out, 256 * 256);
  conv_f2bf(stream, s_W1,   w_w1,  512 * 512);
  conv_f2bf(stream, s_W2,   w_w2,  256 * 512);
  conv_f2bf(stream, c_Wqk,  w_cqk, 256 * 256);
  conv_f2bf(stream, c_Wv,   w_cv,  256 * 256);
  conv_f2bf(stream, c_Wo,   w_co,  256 * 256);
  conv_f2bf(stream, c_W1,   w_cw1, 512 * 512);
  conv_f2bf(stream, c_W2,   w_cw2, 256 * 512);

  // ---- self block (shared weights), run for desc0 and desc1 ----
  auto run_self = [&](const float* desc, const float* enc, float* dst) {
    conv_f2bf(stream, desc, aA, Rows * Dd);
    gemm(stream, aA, w_qkv, s_bqkv, nullptr, g0, Rows, Dd, 3 * Dd);
    rope_pack_kernel<<<dim3(Rows), dim3(256), 0, stream>>>(g0, enc, qb, kb, vb, 0.125f);
    attn_kernel<<<dim3(Nn / 16, BHc), dim3(32), 0, stream>>>(qb, kb, vb, aA);
    gemm(stream, aA, w_out, s_bout, nullptr, msg, Rows, Dd, Dd);
    concat_kernel<<<dim3(Rows * 512 / 256), dim3(256), 0, stream>>>(desc, msg, aA, Rows);
    gemm(stream, aA, w_w1, s_b1, nullptr, g0, Rows, 2 * Dd, 2 * Dd);
    ln_gelu_kernel<<<dim3(Rows), dim3(256), 0, stream>>>(g0, s_g, s_be, aA);
    gemm(stream, aA, w_w2, s_b2, desc, dst, Rows, 2 * Dd, Dd);
  };
  run_self(desc0, enc0, d0s);
  run_self(desc1, enc1, d1s);

  // ---- cross block ----
  conv_f2bf(stream, d0s, db0, Rows * Dd);
  conv_f2bf(stream, d1s, db1, Rows * Dd);

  const float sct = 0.35355339059327373f;   // (HD^-0.5)^0.5
  gemm(stream, db0, w_cqk, c_bqk, nullptr, g0, Rows, Dd, Dd);
  pack_scale_kernel<<<dim3(Rows), dim3(256), 0, stream>>>(g0, qb, sct);   // qk0
  gemm(stream, db1, w_cqk, c_bqk, nullptr, g0, Rows, Dd, Dd);
  pack_scale_kernel<<<dim3(Rows), dim3(256), 0, stream>>>(g0, kb, sct);   // qk1
  gemm(stream, db0, w_cv, c_bv, nullptr, g0, Rows, Dd, Dd);
  pack_scale_kernel<<<dim3(Rows), dim3(256), 0, stream>>>(g0, vb, 1.0f);  // v0
  gemm(stream, db1, w_cv, c_bv, nullptr, g0, Rows, Dd, Dd);
  pack_scale_kernel<<<dim3(Rows), dim3(256), 0, stream>>>(g0, vb2, 1.0f); // v1

  float* out0 = (float*)d_out;
  float* out1 = out0 + (size_t)Rows * Dd;

  auto run_cross_tail = [&](const __bf16* Qp, const __bf16* Kp, const __bf16* Vp,
                            const float* x, float* dst) {
    attn_kernel<<<dim3(Nn / 16, BHc), dim3(32), 0, stream>>>(Qp, Kp, Vp, aA);
    gemm(stream, aA, w_co, c_bo, nullptr, msg, Rows, Dd, Dd);
    concat_kernel<<<dim3(Rows * 512 / 256), dim3(256), 0, stream>>>(x, msg, aA, Rows);
    gemm(stream, aA, w_cw1, c_b1, nullptr, g0, Rows, 2 * Dd, 2 * Dd);
    ln_gelu_kernel<<<dim3(Rows), dim3(256), 0, stream>>>(g0, c_g, c_be, aA);
    gemm(stream, aA, w_cw2, c_b2, x, dst, Rows, 2 * Dd, Dd);
  };
  // m0 = softmax_j(qk0·qk1^T)·v1 ;  m1 = softmax_i(sim)^T·v0 == attention(qk1, qk0, v0)
  run_cross_tail(qb, kb, vb2, d0s, out0);
  run_cross_tail(kb, qb, vb, d1s, out1);
}
